// EdgeAttrGNN_79731772883552
// MI455X (gfx1250) — matrix operand order repaired and checked by reading, not compiled, not run on hardware
//
#include <hip/hip_runtime.h>
#include <hip/hip_bf16.h>
#include <math.h>

// ---------------------------------------------------------------------------
// EdgeAttrGNN (NNConv x2 + mean-pool + MLP head) for MI455X / gfx1250.
//
// Restructure: per-edge weight h_e = relu(ea*w1+b1) @ W2 factors through a
// 64-dim basis r_e, so  msg_e = r_e . T_src + x_src.B2  with per-NODE
// T_n = X[n] (x) W2  computed as ONE dense GEMM per conv via
// v_wmma_f32_16x16x32_bf16.  T (bf16, 82MB) is L2-resident (192MB L2).
//
// Round-2 changes:
//  * W2 packed TRANSPOSED so the WMMA B operand is two contiguous b128 loads
//    per lane (round 1 emitted 16 serialized global_load_u16 per tile).
//  * msg_kernel stages the 8 per-edge r-vectors in LDS (broadcast reads) and
//    prefetches the 4KB T-slice (global_prefetch_b8) before the contraction.
// ---------------------------------------------------------------------------

typedef __bf16 bf16;
typedef __attribute__((ext_vector_type(16))) __bf16 v16bf;
typedef __attribute__((ext_vector_type(8)))  __bf16 bf16x8;
typedef __attribute__((ext_vector_type(8)))  float  v8f;

#define N_NODES  20000
#define N_EDGES  200000
#define HID      32
#define NBASIS   64      // edge-MLP hidden width
#define TCOLS    2048    // NBASIS * HID
#define N_GRAPHS 64
#define N_CLS    10

// ---------------------------------------------------------------- utilities
__global__ void zero_f32_kernel(float* __restrict__ p, int n) {
    int i = blockIdx.x * blockDim.x + threadIdx.x;
    if (i < n) p[i] = 0.0f;
}

__device__ __forceinline__ unsigned hash_u32(unsigned x) {
    x ^= x >> 16; x *= 0x7feb352dU;
    x ^= x >> 15; x *= 0x846ca68bU;
    x ^= x >> 16; return x;
}

// nn2_root is a module-level JAX constant (not an input); synthesize a
// deterministic ~N(0,1)*0.2 stand-in.
__global__ void gen_root2_kernel(float* __restrict__ w) {
    int i = blockIdx.x * blockDim.x + threadIdx.x;
    if (i >= HID * HID) return;
    unsigned h1 = hash_u32(0x9E3779B9u ^ (unsigned)(i * 2 + 1));
    unsigned h2 = hash_u32(0x85EBCA6Bu ^ (unsigned)(i * 2 + 2));
    float u1 = ((float)h1 + 1.0f) * (1.0f / 4294967296.0f);
    float u2 = (float)h2 * (1.0f / 4294967296.0f);
    float z  = sqrtf(-2.0f * logf(u1)) * cosf(6.28318530718f * u2);
    w[i] = 0.2f * z;
}

// ----------------------------------------------------- operand prep kernels
// A operand: node features -> bf16, padded to 32 K-columns.
__global__ void pack_a_kernel(const float* __restrict__ x, bf16* __restrict__ xb,
                              int n_nodes, int in_c) {
    int idx = blockIdx.x * blockDim.x + threadIdx.x;
    if (idx >= n_nodes * 32) return;
    int n = idx >> 5, i = idx & 31;
    xb[idx] = (i < in_c) ? (bf16)x[n * in_c + i] : (bf16)0.0f;
}

// B operand, TRANSPOSED for per-lane-contiguous WMMA loads:
//   BT[col][i] with col = k*32+o (0..2047), i = K-row (0..31)
//   BT[col*32 + i] = w2[k, i*out + o], zero-padded for i >= in_c.
__global__ void pack_w2t_kernel(const float* __restrict__ w2, bf16* __restrict__ BT,
                                int in_c) {
    int idx = blockIdx.x * blockDim.x + threadIdx.x;   // 2048 * 32
    if (idx >= TCOLS * 32) return;
    int col = idx >> 5;
    int i   = idx & 31;
    int k = col >> 5, o = col & 31;
    BT[idx] = (i < in_c) ? (bf16)w2[k * (in_c * HID) + i * HID + o] : (bf16)0.0f;
}

// Per-node bias contraction: u[n,o] = sum_i X[n,i] * b2[i*32+o]
__global__ void u_kernel(const float* __restrict__ xin, const float* __restrict__ b2,
                         float* __restrict__ u, int n_nodes, int in_c) {
    int idx = blockIdx.x * blockDim.x + threadIdx.x;
    if (idx >= n_nodes * 32) return;
    int n = idx >> 5, o = idx & 31;
    float acc = 0.0f;
    for (int i = 0; i < in_c; ++i) acc += xin[n * in_c + i] * b2[i * HID + o];
    u[idx] = acc;
}

// Edge basis: r[e,k] = relu(ea[e]*w1[k] + b1[k])   (bf16)
__global__ void edge_mlp_kernel(const float* __restrict__ ea, const float* __restrict__ w1,
                                const float* __restrict__ b1, bf16* __restrict__ r, int E) {
    int idx = blockIdx.x * blockDim.x + threadIdx.x;
    if (idx >= E * NBASIS) return;
    int e = idx >> 6, k = idx & 63;
    float v = fmaf(ea[e], w1[k], b1[k]);
    r[idx] = (bf16)(v > 0.0f ? v : 0.0f);
}

// --------------------------------------------------------- WMMA dense GEMM
// T[N, 2048] = A[N, 32] @ B[32, 2048]   (bf16 x bf16 -> f32 -> bf16)
// One 16x16x32 WMMA per output tile. Block = 8 waves, one 16-row tile per
// block; wave w covers col-tiles [w*16, w*16+16).  20000 = 16*1250 exact.
// Both operands are built from two contiguous 16B loads per lane.
__global__ __launch_bounds__(256)
void gemm_T_kernel(const bf16* __restrict__ A, const bf16* __restrict__ BT,
                   bf16* __restrict__ T) {
    const int wave   = threadIdx.x >> 5;
    const int lane   = threadIdx.x & 31;
    const int grp    = lane >> 4;       // lane half: 0 or 1
    const int lane16 = lane & 15;
    const int row    = blockIdx.x * 16 + lane16;

    // A layout (16-bit A 16x32, ISA 7.12.2): lane half g holds K-pairs
    //   j<8 -> K = j + 8g ; j>=8 -> K = 8 + j + 8g
    // i.e. two contiguous 8-element runs at K = 8g and K = 16+8g.
    const bf16x8 alo = *(const bf16x8*)(A + (size_t)row * 32 + 8 * grp);
    const bf16x8 ahi = *(const bf16x8*)(A + (size_t)row * 32 + 16 + 8 * grp);
    v16bf a;
#pragma unroll
    for (int j = 0; j < 8; ++j) { a[j] = alo[j]; a[8 + j] = ahi[j]; }

    for (int i = 0; i < 16; ++i) {
        const int ct = wave * 16 + i;         // col tile 0..127
        const int n  = ct * 16 + lane16;      // output column
        // B layout (16-bit B 32x16): same K order as A: runs at K = 8g and K = 16+8g.
        // BT is [col][K] row-major -> two contiguous 8-bf16 runs per lane.
        const bf16x8 blo = *(const bf16x8*)(BT + (size_t)n * 32 + 8 * grp);
        const bf16x8 bhi = *(const bf16x8*)(BT + (size_t)n * 32 + 16 + 8 * grp);
        v16bf b;
#pragma unroll
        for (int j = 0; j < 8; ++j) { b[j] = blo[j]; b[8 + j] = bhi[j]; }

        v8f c = {};
        c = __builtin_amdgcn_wmma_f32_16x16x32_bf16(
                false, a, false, b, (short)0, c, false, false);

        // C/D layout: lane half g, VGPR r -> M = r + 8g, N = lane16.
#pragma unroll
        for (int r = 0; r < 8; ++r) {
            int m = r + 8 * grp;
            T[(size_t)(blockIdx.x * 16 + m) * TCOLS + n] = (bf16)c[r];
        }
    }
}

// ------------------------------------------------------- per-edge messages
// One wave per edge (8 edges/block, grid exactly E/8); lane o computes
// msg[o] = u[src,o] + sum_k r[e,k]*T[src,k,o]; atomic scatter-add to dst.
// r-vectors staged in LDS (broadcast reads); T slice prefetched.
__global__ __launch_bounds__(256)
void msg_kernel(const bf16* __restrict__ r, const bf16* __restrict__ T,
                const float* __restrict__ u, const int* __restrict__ ei,
                float* __restrict__ agg, float* __restrict__ cnt, int E) {
    __shared__ bf16 sr[8 * NBASIS];           // 8 edges x 64 basis = 1KB
    const int tid = threadIdx.x;
    const int e0  = blockIdx.x * 8;

    // Cooperative stage: 512 bf16 = 256 dwords, one per thread (coalesced).
    ((unsigned*)sr)[tid] = ((const unsigned*)r)[(size_t)e0 * 32 + tid];

    const int wave = tid >> 5;
    const int lane = tid & 31;
    const int e    = e0 + wave;
    const int src  = ei[e];
    const int dst  = ei[E + e];

    const bf16* tp = T + (size_t)src * TCOLS + lane;
    // Prefetch the whole 4KB T-slice: 32 lanes x 128B.
    __builtin_prefetch(T + (size_t)src * TCOLS + lane * 64, 0, 0);

    float acc = u[src * 32 + lane];
    __syncthreads();

    const bf16* rp = sr + wave * NBASIS;      // LDS broadcast reads
#pragma unroll 8
    for (int k = 0; k < NBASIS; ++k)
        acc = fmaf((float)rp[k], (float)tp[(size_t)k * 32], acc);

    atomicAdd(&agg[(size_t)dst * 32 + lane], acc);
    if (lane == 0) atomicAdd(&cnt[dst], 1.0f);
}

// ------------------------------------------ combine: mean + root + bias + relu
__global__ __launch_bounds__(256)
void combine_kernel(const float* __restrict__ agg, const float* __restrict__ cnt,
                    const float* __restrict__ xin, const float* __restrict__ root,
                    const float* __restrict__ bias, float* __restrict__ hout,
                    bf16* __restrict__ xb, int n_nodes, int in_c) {
    int n = (int)((blockIdx.x * (size_t)blockDim.x + threadIdx.x) >> 5);
    int o = threadIdx.x & 31;
    if (n >= n_nodes) return;
    float c = cnt[n]; c = c > 1.0f ? c : 1.0f;
    float v = agg[(size_t)n * 32 + o] / c;
    for (int i = 0; i < in_c; ++i)
        v = fmaf(xin[(size_t)n * in_c + i], root[i * HID + o], v);
    if (bias) v += bias[o];
    v = v > 0.0f ? v : 0.0f;
    hout[(size_t)n * 32 + o] = v;
    xb[(size_t)n * 32 + o]   = (bf16)v;   // bf16 A-operand for next conv
}

// ----------------------------------------------------------- pooling + head
__global__ __launch_bounds__(256)
void pool_kernel(const float* __restrict__ h, const int* __restrict__ batch,
                 float* __restrict__ g, float* __restrict__ gcnt, int n_nodes) {
    int n = (int)((blockIdx.x * (size_t)blockDim.x + threadIdx.x) >> 5);
    int o = threadIdx.x & 31;
    if (n >= n_nodes) return;
    int b = batch[n];
    atomicAdd(&g[b * 32 + o], h[(size_t)n * 32 + o]);
    if (o == 0) atomicAdd(&gcnt[b], 1.0f);
}

__global__ void head1_kernel(const float* __restrict__ g, const float* __restrict__ gcnt,
                             const float* __restrict__ w, const float* __restrict__ b,
                             float* __restrict__ t) {
    int idx = blockIdx.x * blockDim.x + threadIdx.x;      // 64*32
    if (idx >= N_GRAPHS * HID) return;
    int gi = idx >> 5, h = idx & 31;
    float c = gcnt[gi]; c = c > 1.0f ? c : 1.0f;
    float acc = b[h];
    for (int j = 0; j < HID; ++j)
        acc = fmaf(g[gi * 32 + j] / c, w[j * HID + h], acc);
    t[idx] = acc > 0.0f ? acc : 0.0f;
}

__global__ void head2_kernel(const float* __restrict__ t, const float* __restrict__ w,
                             const float* __restrict__ b, float* __restrict__ out) {
    int idx = blockIdx.x * blockDim.x + threadIdx.x;      // 64*10
    if (idx >= N_GRAPHS * N_CLS) return;
    int gi = idx / N_CLS, c = idx % N_CLS;
    float acc = b[c];
    for (int h = 0; h < HID; ++h)
        acc = fmaf(t[gi * HID + h], w[h * N_CLS + c], acc);
    out[idx] = acc;
}

// ---------------------------------------------------------------- launcher
extern "C" void kernel_launch(void* const* d_in, const int* in_sizes, int n_in,
                              void* d_out, int out_size, void* d_ws, size_t ws_size,
                              hipStream_t stream) {
    const float* x      = (const float*)d_in[0];
    const float* ea     = (const float*)d_in[1];
    const float* nn1_w1 = (const float*)d_in[2];
    const float* nn1_b1 = (const float*)d_in[3];
    const float* nn1_w2 = (const float*)d_in[4];
    const float* nn1_b2 = (const float*)d_in[5];
    const float* root1  = (const float*)d_in[6];
    const float* bias1  = (const float*)d_in[7];
    const float* nn2_w1 = (const float*)d_in[8];
    const float* nn2_b1 = (const float*)d_in[9];
    const float* nn2_w2 = (const float*)d_in[10];
    const float* nn2_b2 = (const float*)d_in[11];
    const float* lin1_w = (const float*)d_in[12];
    const float* lin1_b = (const float*)d_in[13];
    const float* lin2_w = (const float*)d_in[14];
    const float* lin2_b = (const float*)d_in[15];
    const int*   ei     = (const int*)d_in[16];
    const int*   batch  = (const int*)d_in[17];
    float* out = (float*)d_out;

    // ---- workspace carve-up (256B aligned) -------------------------------
    char* base = (char*)d_ws;
    size_t off = 0;
    auto carve = [&](size_t bytes) -> char* {
        char* p = base + off;
        off = (off + bytes + 255) & ~(size_t)255;
        return p;
    };
    bf16*  r_bf  = (bf16*)carve((size_t)N_EDGES * NBASIS * 2);   // 25.6 MB
    bf16*  T_bf  = (bf16*)carve((size_t)N_NODES * TCOLS * 2);    // 81.9 MB (L2-resident)
    bf16*  xb    = (bf16*)carve((size_t)N_NODES * 32 * 2);
    bf16*  w2t   = (bf16*)carve((size_t)TCOLS * 32 * 2);
    float* u     = (float*)carve((size_t)N_NODES * 32 * 4);
    float* agg   = (float*)carve((size_t)N_NODES * 32 * 4);
    float* cnt   = (float*)carve((size_t)N_NODES * 4);
    float* h1    = (float*)carve((size_t)N_NODES * 32 * 4);
    float* h2    = (float*)carve((size_t)N_NODES * 32 * 4);
    float* root2 = (float*)carve((size_t)HID * HID * 4);
    float* g     = (float*)carve((size_t)N_GRAPHS * 32 * 4);
    float* gcnt  = (float*)carve((size_t)N_GRAPHS * 4);
    float* thead = (float*)carve((size_t)N_GRAPHS * HID * 4);

    const int B = 256;
    auto blocks = [](long n, int b) { return (int)((n + b - 1) / b); };

    gen_root2_kernel<<<blocks(HID * HID, B), B, 0, stream>>>(root2);

    // ======================= Conv 1 (in=16 -> 32) =========================
    pack_a_kernel<<<blocks((long)N_NODES * 32, B), B, 0, stream>>>(x, xb, N_NODES, 16);
    pack_w2t_kernel<<<blocks(TCOLS * 32, B), B, 0, stream>>>(nn1_w2, w2t, 16);
    gemm_T_kernel<<<N_NODES / 16, B, 0, stream>>>(xb, w2t, T_bf);
    u_kernel<<<blocks((long)N_NODES * 32, B), B, 0, stream>>>(x, nn1_b2, u, N_NODES, 16);
    edge_mlp_kernel<<<blocks((long)N_EDGES * NBASIS, B), B, 0, stream>>>(ea, nn1_w1, nn1_b1, r_bf, N_EDGES);
    zero_f32_kernel<<<blocks((long)N_NODES * 32, B), B, 0, stream>>>(agg, N_NODES * 32);
    zero_f32_kernel<<<blocks(N_NODES, B), B, 0, stream>>>(cnt, N_NODES);
    msg_kernel<<<N_EDGES / 8, B, 0, stream>>>(r_bf, T_bf, u, ei, agg, cnt, N_EDGES);
    combine_kernel<<<blocks((long)N_NODES * 32, B), B, 0, stream>>>(
        agg, cnt, x, root1, bias1, h1, xb, N_NODES, 16);

    // ======================= Conv 2 (in=32 -> 32) =========================
    // xb already holds bf16(h1) from combine_kernel.
    pack_w2t_kernel<<<blocks(TCOLS * 32, B), B, 0, stream>>>(nn2_w2, w2t, 32);
    gemm_T_kernel<<<N_NODES / 16, B, 0, stream>>>(xb, w2t, T_bf);
    u_kernel<<<blocks((long)N_NODES * 32, B), B, 0, stream>>>(h1, nn2_b2, u, N_NODES, 32);
    edge_mlp_kernel<<<blocks((long)N_EDGES * NBASIS, B), B, 0, stream>>>(ea, nn2_w1, nn2_b1, r_bf, N_EDGES);
    zero_f32_kernel<<<blocks((long)N_NODES * 32, B), B, 0, stream>>>(agg, N_NODES * 32);
    zero_f32_kernel<<<blocks(N_NODES, B), B, 0, stream>>>(cnt, N_NODES);
    msg_kernel<<<N_EDGES / 8, B, 0, stream>>>(r_bf, T_bf, u, ei, agg, cnt, N_EDGES);
    combine_kernel<<<blocks((long)N_NODES * 32, B), B, 0, stream>>>(
        agg, cnt, h1, root2, (const float*)nullptr, h2, xb, N_NODES, 32);

    // ==================== mean-pool + dense head ==========================
    zero_f32_kernel<<<blocks(N_GRAPHS * 32, B), B, 0, stream>>>(g, N_GRAPHS * 32);
    zero_f32_kernel<<<blocks(N_GRAPHS, B), B, 0, stream>>>(gcnt, N_GRAPHS);
    pool_kernel<<<blocks((long)N_NODES * 32, B), B, 0, stream>>>(h2, batch, g, gcnt, N_NODES);
    head1_kernel<<<blocks(N_GRAPHS * HID, B), B, 0, stream>>>(g, gcnt, lin1_w, lin1_b, thead);
    head2_kernel<<<blocks(N_GRAPHS * N_CLS, B), B, 0, stream>>>(thead, lin2_w, lin2_b, out);
}
